// Attention_48172353192525
// MI455X (gfx1250) — compile-verified, matrix operand order checked by reading, not run on hardware
//
#include <hip/hip_runtime.h>

// ---------------- types ----------------
typedef __bf16 bf16;
typedef __attribute__((ext_vector_type(16))) __bf16 v16bf;
typedef __attribute__((ext_vector_type(8)))  __bf16 v8bf;
typedef __attribute__((ext_vector_type(8)))  float  v8f;

#define D_MODEL   1024
#define NUM_HEADS 16
#define D_HEAD    64
#define SEQ       2048
#define BATCH     2
#define ATTN_C    0.9f

union Frag16 { v16bf v; v8bf h[2]; };

// A/B fragment (16 rows x 32 K, bf16) from a row-major matrix whose rows are
// the 16-dim (M for A, N for B) and whose columns (K) are contiguous.
// ISA 7.12.2: lane l -> row (l&15); K-base ((l>>4)<<3); elems 0..7 = K..K+7,
// elems 8..15 = K+16..K+23.  Two b128 loads (global_load_b128 or ds_load_b128).
__device__ __forceinline__ v16bf load_frag(const bf16* __restrict__ base, int ld,
                                           int row0, int k0) {
  const int lane = threadIdx.x & 31;
  const bf16* p = base + (size_t)(row0 + (lane & 15)) * ld + (k0 + ((lane >> 4) << 3));
  Frag16 f;
  f.h[0] = *(const v8bf*)p;
  f.h[1] = *(const v8bf*)(p + 16);
  return f.v;
}

__device__ __forceinline__ float rmax16(float v) {
  v = fmaxf(v, __shfl_xor(v, 1, 32));
  v = fmaxf(v, __shfl_xor(v, 2, 32));
  v = fmaxf(v, __shfl_xor(v, 4, 32));
  v = fmaxf(v, __shfl_xor(v, 8, 32));
  return v;
}
__device__ __forceinline__ float rsum16(float v) {
  v += __shfl_xor(v, 1, 32);
  v += __shfl_xor(v, 2, 32);
  v += __shfl_xor(v, 4, 32);
  v += __shfl_xor(v, 8, 32);
  return v;
}

// ---------------- fp32 -> bf16 conversion ----------------
__global__ void cvt_bf16_kernel(const float* __restrict__ in, bf16* __restrict__ out, int n) {
  int i = blockIdx.x * blockDim.x + threadIdx.x;
  if (i < n) out[i] = (bf16)in[i];
}

// ---------------- LDS-tiled bf16 WMMA GEMM ----------------
// C[M,N] = A[M,K] @ Bt[N,K]^T.  Block: 256 thr = 8 waves, tile 128(M) x 128(N).
// Wave (w>>1, w&1) owns a 32x64 sub-tile -> 8 f32 accumulators.
// Per 32-wide K-step: cooperative global->LDS staging of A(128x32), B(128x32);
// staging loads issue BEFORE the barrier so they overlap prior WMMA work.
// mode 0: Cf[m*N+n] (fp32)   mode 1: Q/K head layout   mode 2: V^T head layout
__global__ void __launch_bounds__(256)
gemm_bf16_kernel(const bf16* __restrict__ A, const bf16* __restrict__ Bt,
                 int M, int N, int K,
                 float* __restrict__ Cf, bf16* __restrict__ Cb, int mode) {
  __shared__ __align__(16) bf16 As[128 * 32];
  __shared__ __align__(16) bf16 Bs[128 * 32];

  const int tid  = threadIdx.x;
  const int wave = tid >> 5;
  const int lane = tid & 31;
  const int wm   = (wave >> 1) * 32;   // 0,32,64,96
  const int wn   = (wave & 1) * 64;    // 0,64
  const int m0   = blockIdx.x * 128;
  const int n0   = blockIdx.y * 128;

  // staging coordinates: 128 rows x 32 cols = 512 x 16B chunks, 2 per thread
  const int r0 = tid >> 2, c0 = (tid & 3) * 8;          // chunk 0: rows 0..63
  const int r1 = r0 + 64;                               // chunk 1: rows 64..127

  const v8f zero = {0.f, 0.f, 0.f, 0.f, 0.f, 0.f, 0.f, 0.f};
  v8f acc[2][4] = {{zero, zero, zero, zero}, {zero, zero, zero, zero}};

  for (int k0 = 0; k0 < K; k0 += 32) {
    // ---- global -> regs (overlaps previous iteration's WMMA compute) ----
    v8bf ar0 = *(const v8bf*)(A  + (size_t)(m0 + r0) * K + k0 + c0);
    v8bf ar1 = *(const v8bf*)(A  + (size_t)(m0 + r1) * K + k0 + c0);
    v8bf br0 = *(const v8bf*)(Bt + (size_t)(n0 + r0) * K + k0 + c0);
    v8bf br1 = *(const v8bf*)(Bt + (size_t)(n0 + r1) * K + k0 + c0);
    __builtin_prefetch(A  + (size_t)(m0 + r0) * K + k0 + 32, 0, 1);
    __builtin_prefetch(Bt + (size_t)(n0 + r0) * K + k0 + 32, 0, 1);

    __syncthreads();                       // prior frag reads done
    *(v8bf*)(As + r0 * 32 + c0) = ar0;
    *(v8bf*)(As + r1 * 32 + c0) = ar1;
    *(v8bf*)(Bs + r0 * 32 + c0) = br0;
    *(v8bf*)(Bs + r1 * 32 + c0) = br1;
    __syncthreads();                       // tiles visible to all waves

    // ---- compute: frags from LDS, 8 WMMA per wave per K-step ----
    v16bf af[2];
#pragma unroll
    for (int mi = 0; mi < 2; ++mi) af[mi] = load_frag(As, 32, wm + mi * 16, 0);
#pragma unroll
    for (int ni = 0; ni < 4; ++ni) {
      const v16bf bfrag = load_frag(Bs, 32, wn + ni * 16, 0);
#pragma unroll
      for (int mi = 0; mi < 2; ++mi)
        acc[mi][ni] = __builtin_amdgcn_wmma_f32_16x16x32_bf16(
            false, af[mi], false, bfrag, (short)0, acc[mi][ni], false, false);
    }
  }

  const int hi = lane >> 4, col = lane & 15;
#pragma unroll
  for (int mi = 0; mi < 2; ++mi) {
#pragma unroll
    for (int ni = 0; ni < 4; ++ni) {
#pragma unroll
      for (int r = 0; r < 8; ++r) {
        const int m = m0 + wm + mi * 16 + r + 8 * hi;
        const int n = n0 + wn + ni * 16 + col;
        const float val = acc[mi][ni][r];
        if (mode == 0) {
          Cf[(size_t)m * N + n] = val;
        } else {
          const int b = m >> 11, p = m & (SEQ - 1);
          const int i = n >> 6,  h = n & (D_HEAD - 1);
          if (mode == 1)
            Cb[(((size_t)(b * NUM_HEADS + i)) * SEQ + p) * D_HEAD + h] = (bf16)val;
          else
            Cb[(((size_t)(b * NUM_HEADS + i)) * D_HEAD + h) * SEQ + p] = (bf16)val;
        }
      }
    }
  }
}

// ---------------- per-head V column sums (for the +0.1 blend) ----------------
__global__ void vsum_kernel(const bf16* __restrict__ Vt, float* __restrict__ vs) {
  int idx = blockIdx.x * blockDim.x + threadIdx.x;  // (b*H+i)*64 + h, 2048 total
  const bf16* p = Vt + (size_t)idx * SEQ;
  float s = 0.f;
  for (int j = 0; j < SEQ; j += 8) {
    v8bf v = *(const v8bf*)(p + j);
#pragma unroll
    for (int e = 0; e < 8; ++e) s += (float)v[e];
  }
  vs[idx] = s;
}

// ---------------- flash attention core (LDS-tiled) ----------------
// Block: 8 waves, one head, 128 q-rows (16 per wave).  K/V tiles for each
// 32-wide p-block are staged once per block in LDS and shared by all waves.
// z = 0.9 * softmax@V + 0.1 * colsum(V); stored bf16 into z_flat [B,SEQ,H*64].
__global__ void __launch_bounds__(256)
attn_kernel(const bf16* __restrict__ Q, const bf16* __restrict__ Km,
            const bf16* __restrict__ Vt, const float* __restrict__ vs,
            bf16* __restrict__ Z) {
  __shared__ __align__(16) bf16 Ks[32 * 64];        // rows = p (32), ld 64
  __shared__ __align__(16) bf16 Vs[64 * 32];        // rows = h (64), ld 32
  __shared__ __align__(16) bf16 Ps[8][16 * 32];     // per-wave P transpose tile

  const int tid  = threadIdx.x;
  const int wave = tid >> 5;
  const int lane = tid & 31;
  const int bi   = blockIdx.y;                      // b*NUM_HEADS + i
  const int q0   = (blockIdx.x * 8 + wave) * 16;
  const int hi   = lane >> 4, col = lane & 15;

  const bf16* Qh = Q  + (size_t)bi * SEQ * D_HEAD;
  const bf16* Kh = Km + (size_t)bi * SEQ * D_HEAD;
  const bf16* Vh = Vt + (size_t)bi * D_HEAD * SEQ;

  const v16bf qf0 = load_frag(Qh, D_HEAD, q0, 0);
  const v16bf qf1 = load_frag(Qh, D_HEAD, q0, 32);

  // staging coords: K tile 32x64 (8 chunks/row), V tile 64x32 (4 chunks/row)
  const int kr = tid >> 3, kc = (tid & 7) * 8;      // 256 chunks
  const int vr = tid >> 2, vc = (tid & 3) * 8;      // 256 chunks

  const v8f zero = {0.f, 0.f, 0.f, 0.f, 0.f, 0.f, 0.f, 0.f};
  v8f acc[4] = {zero, zero, zero, zero};
  float mrow[8], lrow[8];
#pragma unroll
  for (int r = 0; r < 8; ++r) { mrow[r] = -1e30f; lrow[r] = 0.f; }

  for (int p0 = 0; p0 < SEQ; p0 += 32) {
    // ---- global -> regs (overlaps previous iteration's compute) ----
    v8bf kreg = *(const v8bf*)(Kh + (size_t)(p0 + kr) * D_HEAD + kc);
    v8bf vreg = *(const v8bf*)(Vh + (size_t)vr * SEQ + p0 + vc);

    __syncthreads();
    *(v8bf*)(Ks + kr * 64 + kc) = kreg;
    *(v8bf*)(Vs + vr * 32 + vc) = vreg;
    __syncthreads();

    // ---- S = (Q @ K^T) / 8: two 16x16 tiles, operands from LDS ----
    v8f s0 = zero, s1 = zero;
    s0 = __builtin_amdgcn_wmma_f32_16x16x32_bf16(false, qf0, false,
             load_frag(Ks, 64, 0, 0),   (short)0, s0, false, false);
    s0 = __builtin_amdgcn_wmma_f32_16x16x32_bf16(false, qf1, false,
             load_frag(Ks, 64, 0, 32),  (short)0, s0, false, false);
    s1 = __builtin_amdgcn_wmma_f32_16x16x32_bf16(false, qf0, false,
             load_frag(Ks, 64, 16, 0),  (short)0, s1, false, false);
    s1 = __builtin_amdgcn_wmma_f32_16x16x32_bf16(false, qf1, false,
             load_frag(Ks, 64, 16, 32), (short)0, s1, false, false);

    // ---- online softmax update ----
    float corr[8];
#pragma unroll
    for (int r = 0; r < 8; ++r) {
      const float a0 = s0[r] * 0.125f;
      const float a1 = s1[r] * 0.125f;
      const float mnew = fmaxf(mrow[r], rmax16(fmaxf(a0, a1)));
      corr[r] = __expf(mrow[r] - mnew);
      const float e0 = __expf(a0 - mnew);
      const float e1 = __expf(a1 - mnew);
      const int row = r + 8 * hi;
      Ps[wave][row * 32 + col]      = (bf16)e0;   // C-layout -> row-major LDS
      Ps[wave][row * 32 + 16 + col] = (bf16)e1;
      lrow[r] = lrow[r] * corr[r] + rsum16(e0 + e1);
      mrow[r] = mnew;
    }
#pragma unroll
    for (int t = 0; t < 4; ++t)
#pragma unroll
      for (int r = 0; r < 8; ++r) acc[t][r] *= corr[r];

    // ---- P (A-frag via per-wave LDS transpose) @ V ----
    const v16bf pf = load_frag(&Ps[wave][0], 32, 0, 0);
#pragma unroll
    for (int t = 0; t < 4; ++t) {
      const v16bf vb = load_frag(Vs, 32, t * 16, 0);   // rows = h, K = p
      acc[t] = __builtin_amdgcn_wmma_f32_16x16x32_bf16(
          false, pf, false, vb, (short)0, acc[t], false, false);
    }
  }

  // ---- epilogue: z = 0.9 * O/l + 0.1 * colsum(V) ----
  const int b = bi >> 4, ihead = bi & (NUM_HEADS - 1);
#pragma unroll
  for (int t = 0; t < 4; ++t) {
#pragma unroll
    for (int r = 0; r < 8; ++r) {
      const int qrow = q0 + r + 8 * hi;
      const int h = t * 16 + col;
      const float z = ATTN_C * (acc[t][r] / lrow[r]) +
                      (1.0f - ATTN_C) * vs[bi * D_HEAD + h];
      Z[((size_t)(b * SEQ + qrow)) * (NUM_HEADS * D_HEAD) + ihead * D_HEAD + h] = (bf16)z;
    }
  }
}

// ---------------- host side ----------------
extern "C" void kernel_launch(void* const* d_in, const int* in_sizes, int n_in,
                              void* d_out, int out_size, void* d_ws, size_t ws_size,
                              hipStream_t stream) {
  (void)in_sizes; (void)n_in; (void)out_size; (void)ws_size;
  const float* x   = (const float*)d_in[0];
  const float* W_K = (const float*)d_in[1];
  const float* W_Q = (const float*)d_in[2];
  const float* W_V = (const float*)d_in[3];
  const float* W_O = (const float*)d_in[4];
  float* out = (float*)d_out;

  const size_t nX = (size_t)BATCH * SEQ * D_MODEL;                 // 4M
  const size_t nW = (size_t)NUM_HEADS * D_HEAD * D_MODEL;          // 1M
  const size_t nQ = (size_t)BATCH * NUM_HEADS * SEQ * D_HEAD;      // 4M

  char* w = (char*)d_ws;
  bf16* xb  = (bf16*)w; w += nX * sizeof(bf16);
  bf16* wkb = (bf16*)w; w += nW * sizeof(bf16);
  bf16* wqb = (bf16*)w; w += nW * sizeof(bf16);
  bf16* wvb = (bf16*)w; w += nW * sizeof(bf16);
  bf16* wob = (bf16*)w; w += nW * sizeof(bf16);
  bf16* Qb  = (bf16*)w; w += nQ * sizeof(bf16);
  bf16* Kb  = (bf16*)w; w += nQ * sizeof(bf16);
  bf16* Vtb = (bf16*)w; w += nQ * sizeof(bf16);
  float* vs = (float*)w; w += (size_t)BATCH * NUM_HEADS * D_HEAD * sizeof(float);
  bf16* Zb  = (bf16*)w;

  // 1) convert everything to bf16
  cvt_bf16_kernel<<<dim3((nX + 255) / 256), dim3(256), 0, stream>>>(x,   xb,  (int)nX);
  cvt_bf16_kernel<<<dim3((nW + 255) / 256), dim3(256), 0, stream>>>(W_K, wkb, (int)nW);
  cvt_bf16_kernel<<<dim3((nW + 255) / 256), dim3(256), 0, stream>>>(W_Q, wqb, (int)nW);
  cvt_bf16_kernel<<<dim3((nW + 255) / 256), dim3(256), 0, stream>>>(W_V, wvb, (int)nW);
  cvt_bf16_kernel<<<dim3((nW + 255) / 256), dim3(256), 0, stream>>>(W_O, wob, (int)nW);

  // 2) QKV projections: [4096,1024] @ [1024,1024]^T, block tile 128x128
  const int M = BATCH * SEQ, N = D_MODEL, K = D_MODEL;
  dim3 ggrid(M / 128, N / 128), gblk(256);
  gemm_bf16_kernel<<<ggrid, gblk, 0, stream>>>(xb, wqb, M, N, K, nullptr, Qb,  1);
  gemm_bf16_kernel<<<ggrid, gblk, 0, stream>>>(xb, wkb, M, N, K, nullptr, Kb,  1);
  gemm_bf16_kernel<<<ggrid, gblk, 0, stream>>>(xb, wvb, M, N, K, nullptr, Vtb, 2);

  // 3) V column sums for the post-softmax constant blend
  vsum_kernel<<<dim3((BATCH * NUM_HEADS * D_HEAD) / 256), dim3(256), 0, stream>>>(Vtb, vs);

  // 4) flash attention -> z_flat (bf16)
  attn_kernel<<<dim3(SEQ / 128, BATCH * NUM_HEADS), dim3(256), 0, stream>>>(Qb, Kb, Vtb, vs, Zb);

  // 5) output projection: out = z_flat @ W_O^T (fp32 result)
  gemm_bf16_kernel<<<ggrid, gblk, 0, stream>>>(Zb, wob, M, N, K, out, nullptr, 0);
}
